// EdgeGen_69217692942520
// MI455X (gfx1250) — matrix operand order repaired and checked by reading, not compiled, not run on hardware
//
#include <hip/hip_runtime.h>

#define NN 8192
#define DD 512
#define PP 16
#define KTOT (PP * DD)     // 8192
#define EPSF 0.1f
#define TOPK 64

typedef __attribute__((ext_vector_type(16))) __bf16 v16bf;
typedef __attribute__((ext_vector_type(8)))  float  v8f;
typedef __attribute__((ext_vector_type(4)))  unsigned int v4u;
typedef __attribute__((ext_vector_type(8)))  int v8i_;
typedef __attribute__((ext_vector_type(4)))  int v4i_;

#define HAVE_TDM __has_builtin(__builtin_amdgcn_tensor_load_to_lds)
#if __has_include(<hip/amd_detail/amd_gfx1250_TDM.h>)
#define TDM_SIX_ARG 1
#else
#define TDM_SIX_ARG 0
#endif

__device__ __forceinline__ unsigned f2bf(float f) {
    unsigned u = __float_as_uint(f);
    u += 0x7FFFu + ((u >> 16) & 1u);   // round-to-nearest-even
    return u >> 16;
}

// ---------------------------------------------------------------------------
// Kernel 1: X[n, p*512+d] = bf16( nf[n,d]*w[p,d] / max(||ctx_{p,n}||, 1e-12) )
// One block per node n, one wave per perspective p (16 waves = 512 threads).
// ---------------------------------------------------------------------------
__global__ __launch_bounds__(512)
void prep_kernel(const float* __restrict__ nf, const float* __restrict__ w,
                 unsigned short* __restrict__ X)
{
    const int n    = blockIdx.x;
    const int p    = threadIdx.x >> 5;
    const int lane = threadIdx.x & 31;

    const float* nrow = nf + (size_t)n * DD;
    const float* wrow = w  + (size_t)p * DD;

    float va[8], vb[8];
    float ss = 0.f;
#pragma unroll
    for (int j = 0; j < 8; ++j) {
        int d = 2 * lane + 64 * j;           // pairs, coalesced
        float x0 = nrow[d]     * wrow[d];
        float x1 = nrow[d + 1] * wrow[d + 1];
        va[j] = x0; vb[j] = x1;
        ss += x0 * x0 + x1 * x1;
    }
#pragma unroll
    for (int off = 16; off > 0; off >>= 1)
        ss += __shfl_xor(ss, off);

    float scale = 1.f / fmaxf(sqrtf(ss), 1e-12f);

    unsigned* dst = (unsigned*)(X + (size_t)n * KTOT + (size_t)p * DD);
#pragma unroll
    for (int j = 0; j < 8; ++j) {
        int d = 2 * lane + 64 * j;
        unsigned lo = f2bf(va[j] * scale);
        unsigned hi = f2bf(vb[j] * scale);
        dst[d >> 1] = lo | (hi << 16);       // packed 2 x bf16, coalesced
    }
}

// ---------------------------------------------------------------------------
// Kernel 2: adj = relu_eps( (X X^T) / 16 )  via v_wmma_f32_16x16x32_bf16.
// Block tile 128x128, 4 waves, each wave 64x64 = 4x4 WMMA accumulators.
// Staging: TDM tensor_load_to_lds, double-buffered, padded LDS stride 72
// halves (TDM pad: 4 DWORDs after every 32 DWORDs) -> 16B-aligned ds reads.
// ---------------------------------------------------------------------------
#define BM   128
#define BK   64
#define LDST 72                       // padded halves per LDS tile row
#define TILE_H   (BM * LDST)          // 9216 halves per tile
#define TILE_B   (TILE_H * 2)         // 18432 bytes per tile
#define GEMM_LDS (4 * TILE_B)         // 73728 bytes: {A,B} x 2 stages

#if HAVE_TDM
// One TDM descriptor: load a BM x BK bf16 tile (row stride KTOT elements)
// from global 'gaddr' into LDS 'lds_addr', LDS row stride LDST halves.
__device__ __forceinline__ void tdm_load_tile(unsigned lds_addr,
                                              unsigned long long gaddr)
{
    v4u g0;
    g0.x = 1u;                                            // count=1 (valid)
    g0.y = lds_addr;                                      // LDS byte address
    g0.z = (unsigned)(gaddr & 0xFFFFFFFFull);             // global_addr[31:0]
    g0.w = (unsigned)((gaddr >> 32) & 0x01FFFFFFull)      // global_addr[56:32]
         | (2u << 30);                                    // type=2 ("image")

    v8i_ g1;
    g1[0] = (int)((1u << 16)      // data_size = 2 bytes
                | (1u << 20)      // pad_enable
                | (4u << 22)      // pad_interval: 32 DWORDs (=128B = one row)
                | (3u << 25));    // pad_amount:   4 DWORDs (=16B) -> stride 72
    g1[1] = (int)(((unsigned)KTOT & 0xFFFFu) << 16);      // tensor_dim0 lo16
    g1[2] = (int)((((unsigned)KTOT >> 16) & 0xFFFFu)      // tensor_dim0 hi16
                | (((unsigned)NN & 0xFFFFu) << 16));      // tensor_dim1 lo16
    g1[3] = (int)((((unsigned)NN >> 16) & 0xFFFFu)        // tensor_dim1 hi16
                | ((unsigned)BK << 16));                  // tile_dim0 = 64
    g1[4] = (int)BM;                                      // tile_dim1=128, tile_dim2=0
    g1[5] = (int)KTOT;                                    // tensor_dim0_stride lo32
    g1[6] = 0;                                            // stride hi | dim1_stride lo
    g1[7] = 0;

    v4i_ z4 = {0, 0, 0, 0};
#if TDM_SIX_ARG
    v8i_ z8 = {0, 0, 0, 0, 0, 0, 0, 0};
    __builtin_amdgcn_tensor_load_to_lds(g0, g1, z4, z4, z8, 0);
#else
    __builtin_amdgcn_tensor_load_to_lds(g0, g1, z4, z4, 0);
#endif
}
#endif // HAVE_TDM

union FragU { uint4 u[2]; v16bf v; };

// 2 kk-steps x 4x4 WMMA on one staged {A,B} tile pair.
__device__ __forceinline__ void mma_stage(const unsigned short* __restrict__ ldsA,
                                          const unsigned short* __restrict__ ldsB,
                                          int wi, int wj, int lsub, int hkA,
                                          int ksubB, v8f acc[4][4])
{
#pragma unroll
    for (int kk = 0; kk < BK; kk += 32) {
        FragU afr[4], bfr[4];
#pragma unroll
        for (int a = 0; a < 4; ++a) {
            // A 16x32 bf16: lane m = lsub; elems 0..7 -> K[hkA..hkA+7],
            //                              elems 8..15 -> K[16+hkA..]
            const unsigned short* p = ldsA + (wi + a * 16 + lsub) * LDST + kk;
            afr[a].u[0] = *(const uint4*)(p + hkA);
            afr[a].u[1] = *(const uint4*)(p + 16 + hkA);
        }
#pragma unroll
        for (int b = 0; b < 4; ++b) {
            // B 32x16 bf16: lane n = lsub; elems 0..7 -> K[hkA..hkA+7], elems 8..15 -> K[16+hkA..]
            const unsigned short* p = ldsB + (wj + b * 16 + lsub) * LDST + kk;
            bfr[b].u[0] = *(const uint4*)(p + hkA);
            bfr[b].u[1] = *(const uint4*)(p + 16 + hkA);
        }
#pragma unroll
        for (int a = 0; a < 4; ++a)
#pragma unroll
            for (int b = 0; b < 4; ++b)
                acc[a][b] = __builtin_amdgcn_wmma_f32_16x16x32_bf16(
                    false, afr[a].v, false, bfr[b].v,
                    (short)0, acc[a][b], false, false);
    }
}

__global__ __launch_bounds__(128)
void wmma_gemm_kernel(const unsigned short* __restrict__ Xu, float* __restrict__ out)
{
    extern __shared__ __align__(16) unsigned short dynlds[];

    const int tid  = threadIdx.x;
    const int i0   = blockIdx.y * BM;     // output rows  (A = X rows i)
    const int j0   = blockIdx.x * BM;     // output cols  (B = X rows j)
    const int wv   = tid >> 5;
    const int lane = tid & 31;
    const int wi   = (wv >> 1) * 64;      // wave tile origin in block (rows)
    const int wj   = (wv & 1)  * 64;      //                            (cols)

    // CDNA5 16-bit WMMA lane mapping
    const int lsub  = lane & 15;          // A: row M; B: col N; D: col N
    const int lhalf = lane >> 4;          // lane group 0/1
    const int hkA   = lhalf * 8;          // A frag K sub-offset
    const int ksubB = lhalf * 16;         // unused; B frag uses hkA, same as A
    const int lrow8 = lhalf * 8;          // D frag row offset

    const v8f vz = {0.f, 0.f, 0.f, 0.f, 0.f, 0.f, 0.f, 0.f};
    v8f acc[4][4];
#pragma unroll
    for (int a = 0; a < 4; ++a)
#pragma unroll
        for (int b = 0; b < 4; ++b) acc[a][b] = vz;

#if HAVE_TDM
    // --- TDM double-buffered pipeline: wave 0 is the producer -------------
    const unsigned long long xbase = (unsigned long long)(size_t)Xu;
    const unsigned ldsbase = (unsigned)(size_t)dynlds;   // LDS aperture low32
    const int S = KTOT / BK;

    if (wv == 0) {
        tdm_load_tile(ldsbase,          xbase + (size_t)i0 * KTOT * 2);
        tdm_load_tile(ldsbase + TILE_B, xbase + (size_t)j0 * KTOT * 2);
    }
    for (int s = 0; s < S; ++s) {
        const int par = s & 1;
        if (wv == 0) __builtin_amdgcn_s_wait_tensorcnt(0);  // cur stage landed
        __syncthreads();                                    // all see it; prev bufs free
        if (wv == 0 && s + 1 < S) {                         // issue next stage
            unsigned nb = (unsigned)(par ^ 1) * (2u * TILE_B);
            size_t k0n = (size_t)(s + 1) * BK;
            tdm_load_tile(ldsbase + nb,
                          xbase + ((size_t)i0 * KTOT + k0n) * 2);
            tdm_load_tile(ldsbase + nb + TILE_B,
                          xbase + ((size_t)j0 * KTOT + k0n) * 2);
        }
        const unsigned short* ldsA = dynlds + par * (2 * TILE_H);
        mma_stage(ldsA, ldsA + TILE_H, wi, wj, lsub, hkA, ksubB, acc);
    }
#else
    // --- Fallback: synchronous cooperative global->LDS staging ------------
    unsigned short* ldsA = dynlds;
    unsigned short* ldsB = dynlds + TILE_H;
    for (int k0 = 0; k0 < KTOT; k0 += BK) {
        __syncthreads();
#pragma unroll
        for (int it = 0; it < 16; ++it) {
            int idx  = it * 128 + tid;          // 0..2047
            int isB  = idx >> 10;
            int r    = (idx & 1023) >> 3;       // row in tile
            int c    = idx & 7;                 // 16-byte chunk in row
            int grow = (isB ? j0 : i0) + r;
            const unsigned short* gp = Xu + (size_t)grow * KTOT + k0 + c * 8;
            uint4 v = *(const uint4*)gp;
            if (k0 + BK < KTOT)
                __builtin_prefetch((const void*)(gp + BK), 0, 1);
            *(uint4*)((isB ? ldsB : ldsA) + r * LDST + c * 8) = v;
        }
        __syncthreads();
        mma_stage(ldsA, ldsB, wi, wj, lsub, hkA, ksubB, acc);
    }
#endif

    // Epilogue: /16, epsilon threshold, fp32 store.
    const float inv = 1.f / (float)PP;
#pragma unroll
    for (int a = 0; a < 4; ++a) {
#pragma unroll
        for (int b = 0; b < 4; ++b) {
            int gi = i0 + wi + a * 16 + lrow8;
            int gj = j0 + wj + b * 16 + lsub;
            float* o = out + (size_t)gi * NN + gj;
#pragma unroll
            for (int i = 0; i < 8; ++i) {
                float v = acc[a][b][i] * inv;
                o[(size_t)i * NN] = (v > EPSF) ? v : 0.f;
            }
        }
    }
}

// ---------------------------------------------------------------------------
// Kernel 3: per-row top-64 keep. Radix-select (4 x 8-bit MSB-first passes) on
// float bit patterns (all candidates > 0, so uint order == float order).
// One 256-thread block per row; row cached in LDS; rewrite in place.
// ---------------------------------------------------------------------------
__global__ __launch_bounds__(256)
void topk_kernel(float* __restrict__ out)
{
    __shared__ unsigned bits[NN];      // 32 KB
    __shared__ unsigned hist[256];
    __shared__ unsigned s_prefix, s_mask, s_kneed;

    const int row = blockIdx.x;
    float* rptr = out + (size_t)row * NN;

    for (int j = threadIdx.x; j < NN; j += 256)
        bits[j] = __float_as_uint(rptr[j]);
    if (threadIdx.x == 0) { s_prefix = 0u; s_mask = 0u; s_kneed = TOPK; }
    __syncthreads();

    for (int shift = 24; shift >= 0; shift -= 8) {
        hist[threadIdx.x] = 0u;
        __syncthreads();
        unsigned mask = s_mask, pref = s_prefix;
        for (int j = threadIdx.x; j < NN; j += 256) {
            unsigned u = bits[j];
            if (u != 0u && (u & mask) == pref)
                atomicAdd(&hist[(u >> shift) & 255u], 1u);
        }
        __syncthreads();
        if (threadIdx.x == 0) {
            unsigned k = s_kneed, cum = 0u;
            int sel = 0;
            for (int b = 255; b >= 0; --b) {
                unsigned c = hist[b];
                if (cum + c >= k) { sel = b; s_kneed = k - cum; break; }
                cum += c;
            }
            s_prefix = pref | ((unsigned)sel << (unsigned)shift);
            s_mask   = mask | (0xFFu << (unsigned)shift);
        }
        __syncthreads();
    }

    const unsigned thr = s_prefix;   // bit pattern of the 64th-largest value
    for (int j = threadIdx.x; j < NN; j += 256) {
        unsigned u = bits[j];
        rptr[j] = (u != 0u && u >= thr) ? __uint_as_float(u) : 0.f;
    }
}

// ---------------------------------------------------------------------------
extern "C" void kernel_launch(void* const* d_in, const int* in_sizes, int n_in,
                              void* d_out, int out_size, void* d_ws, size_t ws_size,
                              hipStream_t stream)
{
    (void)in_sizes; (void)n_in; (void)out_size; (void)ws_size;
    const float* nf = (const float*)d_in[0];   // [8192, 512] fp32
    const float* w  = (const float*)d_in[1];   // [16, 512]   fp32
    float* out = (float*)d_out;                // [8192, 8192] fp32
    unsigned short* X = (unsigned short*)d_ws; // [8192, 8192] bf16 = 128 MB

    prep_kernel<<<dim3(NN), 512, 0, stream>>>(nf, w, X);
    wmma_gemm_kernel<<<dim3(NN / BM, NN / BM), 128, GEMM_LDS, stream>>>(X, out);
    topk_kernel<<<dim3(NN), 256, 0, stream>>>(out);
}
